// CrossAttention_69286412419541
// MI455X (gfx1250) — matrix operand order repaired and checked by reading, not compiled, not run on hardware
//
#include <hip/hip_runtime.h>
#include <hip/hip_bf16.h>

typedef __bf16 bf16_t;
typedef __attribute__((ext_vector_type(16))) __bf16 v16bf;
typedef __attribute__((ext_vector_type(8)))  __bf16 v8bf;
typedef __attribute__((ext_vector_type(4)))  __bf16 v4bf;
typedef __attribute__((ext_vector_type(8)))  float  v8f;

// ---------------------------------------------------------------------------
// Fragment loaders (wave32, V_WMMA_F32_16X16X32_BF16 contiguous k order (k = 16*(lane/16) + s) in A and B)
// ---------------------------------------------------------------------------
// A: 16x32 (MxK) bf16 row-major. Lane L<16 holds row L, K = 0..15;
// lane L>=16 holds row L-16, K = 16..31 (same contiguous order as B).
__device__ __forceinline__ v16bf load_frag_a(const bf16_t* __restrict__ A0, int lda, int lane) {
    const int row  = lane & 15;
    const int kOff = (lane & 16) ? 16 : 0;
    return *reinterpret_cast<const v16bf*>(A0 + (size_t)row * lda + kOff);
}

// B: 32x16 (KxN) bf16. Memory layout = "column major B" i.e. Bcol[n][k] row-major
// with stride ldb. Lane L holds column n = L&15; lanes 0-15 hold K=0..15,
// lanes 16-31 hold K=16..31 -> one contiguous 32B load per lane.
__device__ __forceinline__ v16bf load_frag_b(const bf16_t* __restrict__ B0, int ldb, int lane) {
    const int n = lane & 15;
    const int k = (lane & 16) ? 16 : 0;
    return *reinterpret_cast<const v16bf*>(B0 + (size_t)n * ldb + k);
}

// ---------------------------------------------------------------------------
// Generic bf16 WMMA GEMM:  C[M,N] = scale * (A[M,K] * Bcol[N,K]^T + bias[N])
// One wave computes a 64x64 tile: per 32-wide K step, 4 A fragments + 4 B
// fragments feed 16 WMMAs (each fragment reused 4x -> 16 B loaded per WMMA).
// OUT_MODE 0: f32 row-major   OUT_MODE 1: bf16 row-major
// OUT_MODE 2: bf16, transposed + per-batch (Vt[b][n][s]) store
// ---------------------------------------------------------------------------
template <int OUT_MODE>
__global__ __launch_bounds__(256)
void gemm_bf16_kernel(const bf16_t* __restrict__ A, int lda,
                      const bf16_t* __restrict__ B, int ldb,
                      void* __restrict__ C, int ldc,
                      const float* __restrict__ bias, float scale,
                      int M, int N, int K,
                      int tRowsPerBatch, long long cBatchStride)
{
    const int lane = threadIdx.x & 31;
    const int wave = threadIdx.x >> 5;
    const int nGroups    = N >> 6;                 // 64-wide column groups
    const int totalTiles = (M >> 6) * nGroups;     // 64-row tiles
    const int tile = blockIdx.x * 8 + wave;
    if (tile >= totalTiles) return;
    const int mTile = tile / nGroups;
    const int ng    = tile % nGroups;
    const int row0  = mTile << 6;
    const int col0  = ng << 6;

    v8f acc[4][4] = {};
    const bf16_t* Ap = A + (size_t)row0 * lda;
    const bf16_t* Bp = B + (size_t)col0 * ldb;

    for (int k = 0; k < K; k += 32) {
        // Unconditional speculative prefetch of the next A panel
        // (global_prefetch_b8; harmless one panel past the end).
        __builtin_prefetch(Ap + k + 32, 0, 3);
        v16bf a[4];
#pragma unroll
        for (int r = 0; r < 4; ++r)
            a[r] = load_frag_a(Ap + (size_t)(r * 16) * lda + k, lda, lane);
#pragma unroll
        for (int j = 0; j < 4; ++j) {
            v16bf b = load_frag_b(Bp + (size_t)(j * 16) * ldb + k, ldb, lane);
#pragma unroll
            for (int r = 0; r < 4; ++r) {
                acc[r][j] = __builtin_amdgcn_wmma_f32_16x16x32_bf16(
                    false, a[r], false, b, (short)0, acc[r][j], false, false);
            }
        }
    }

    // C fragment layout: element i -> row = rowBase + i + (lane>=16 ? 8 : 0),
    //                                 col = col0 + 16*j + (lane & 15)
    const int rOff = (lane & 16) ? 8 : 0;
    const int cc   = lane & 15;
#pragma unroll
    for (int r = 0; r < 4; ++r) {
        const int rowBase = row0 + r * 16 + rOff;
#pragma unroll
        for (int j = 0; j < 4; ++j) {
            const int col = col0 + j * 16 + cc;
            const float bv = bias ? bias[col] : 0.0f;
            if (OUT_MODE == 0) {
                float* Cf = (float*)C;
#pragma unroll
                for (int i = 0; i < 8; ++i) {
                    const int row = rowBase + i;
                    Cf[(size_t)row * ldc + col] = (acc[r][j][i] + bv) * scale;
                }
            } else if (OUT_MODE == 1) {
                bf16_t* Cb = (bf16_t*)C;
#pragma unroll
                for (int i = 0; i < 8; ++i) {
                    const int row = rowBase + i;
                    Cb[(size_t)row * ldc + col] = (bf16_t)((acc[r][j][i] + bv) * scale);
                }
            } else {
                // Transposed per-batch store: global row = b*tRowsPerBatch + s
                bf16_t* Cb = (bf16_t*)C;
                const int b  = row0 / tRowsPerBatch;
                const int s0 = (rowBase % tRowsPerBatch);
                v8bf v;
#pragma unroll
                for (int i = 0; i < 8; ++i) v[i] = (bf16_t)((acc[r][j][i] + bv) * scale);
                *reinterpret_cast<v8bf*>(Cb + (size_t)b * cBatchStride
                                            + (size_t)col * ldc + s0) = v;
            }
        }
    }
}

// ---------------------------------------------------------------------------
// Row softmax: S(f32)[row][0..n) -> P(bf16). One 256-thread block per row.
// ---------------------------------------------------------------------------
__global__ __launch_bounds__(256)
void softmax_bf16_kernel(const float* __restrict__ S, bf16_t* __restrict__ P, int n)
{
    __shared__ float red[256];
    const int row = blockIdx.x;
    const int t   = threadIdx.x;
    const float* s = S + (size_t)row * n;
    bf16_t*      p = P + (size_t)row * n;

    float m = -3.402823466e38f;
    for (int i = t; i < n; i += 256) m = fmaxf(m, s[i]);
    red[t] = m; __syncthreads();
    for (int off = 128; off > 0; off >>= 1) {
        if (t < off) red[t] = fmaxf(red[t], red[t + off]);
        __syncthreads();
    }
    m = red[0]; __syncthreads();

    float sum = 0.0f;
    for (int i = t; i < n; i += 256) sum += __expf(s[i] - m);
    red[t] = sum; __syncthreads();
    for (int off = 128; off > 0; off >>= 1) {
        if (t < off) red[t] += red[t + off];
        __syncthreads();
    }
    const float inv = 1.0f / red[0];
    for (int i = t; i < n; i += 256) p[i] = (bf16_t)(__expf(s[i] - m) * inv);
}

// ---------------------------------------------------------------------------
// fp32 -> bf16 convert, 4 elements/thread (n divisible by 1024 here)
// ---------------------------------------------------------------------------
__global__ __launch_bounds__(256)
void cvt_f32_to_bf16_kernel(const float* __restrict__ in, bf16_t* __restrict__ out, int n)
{
    const int i = (blockIdx.x * 256 + threadIdx.x) * 4;
    if (i + 3 < n) {
        const float4 v = *reinterpret_cast<const float4*>(in + i);
        v4bf o;
        o[0] = (bf16_t)v.x; o[1] = (bf16_t)v.y; o[2] = (bf16_t)v.z; o[3] = (bf16_t)v.w;
        *reinterpret_cast<v4bf*>(out + i) = o;
    }
}

// ---------------------------------------------------------------------------
extern "C" void kernel_launch(void* const* d_in, const int* in_sizes, int n_in,
                              void* d_out, int out_size, void* d_ws, size_t ws_size,
                              hipStream_t stream)
{
    (void)in_sizes; (void)n_in; (void)out_size; (void)ws_size;

    constexpr int B  = 8;
    constexpr int T  = 2048;   // Tq == Tk
    constexpr int D  = 1024;
    constexpr int MA = B * T;  // 16384 flattened rows

    const float* target = (const float*)d_in[0];
    const float* source = (const float*)d_in[1];
    const float* Wq = (const float*)d_in[2];  const float* bq = (const float*)d_in[3];
    const float* Wk = (const float*)d_in[4];  const float* bk = (const float*)d_in[5];
    const float* Wv = (const float*)d_in[6];  const float* bv = (const float*)d_in[7];
    const float* Wo = (const float*)d_in[8];  const float* bo = (const float*)d_in[9];
    float* out = (float*)d_out;

    // ---- workspace carve-up (all 256B aligned) ----
    char* w = (char*)d_ws;
    auto carve = [&](size_t bytes) -> void* {
        void* p = (void*)w;
        w += (bytes + 255) & ~(size_t)255;
        return p;
    };
    bf16_t* tgt_b = (bf16_t*)carve((size_t)MA * D * 2);   // 32 MB
    bf16_t* src_b = (bf16_t*)carve((size_t)MA * D * 2);   // 32 MB
    bf16_t* Wq_b  = (bf16_t*)carve((size_t)D * D * 2);
    bf16_t* Wk_b  = (bf16_t*)carve((size_t)D * D * 2);
    bf16_t* Wv_b  = (bf16_t*)carve((size_t)D * D * 2);
    bf16_t* Wo_b  = (bf16_t*)carve((size_t)D * D * 2);
    bf16_t* Qb    = (bf16_t*)carve((size_t)MA * D * 2);   // 32 MB (pre-scaled by 1/sqrt(D))
    bf16_t* Kb    = (bf16_t*)carve((size_t)MA * D * 2);   // 32 MB
    bf16_t* Vt    = (bf16_t*)carve((size_t)B * D * T * 2);// 32 MB, [b][e][s]
    float*  Sb    = (float*) carve((size_t)T * T * 4);    // 16 MB, one batch, reused
    bf16_t* Pb    = (bf16_t*)carve((size_t)T * T * 2);    //  8 MB, one batch, reused
    bf16_t* Ob    = (bf16_t*)carve((size_t)MA * D * 2);   // 32 MB

    // ---- 1. fp32 -> bf16 converts ----
    cvt_f32_to_bf16_kernel<<<(MA * D) / 1024, 256, 0, stream>>>(target, tgt_b, MA * D);
    cvt_f32_to_bf16_kernel<<<(MA * D) / 1024, 256, 0, stream>>>(source, src_b, MA * D);
    cvt_f32_to_bf16_kernel<<<(D * D) / 1024, 256, 0, stream>>>(Wq, Wq_b, D * D);
    cvt_f32_to_bf16_kernel<<<(D * D) / 1024, 256, 0, stream>>>(Wk, Wk_b, D * D);
    cvt_f32_to_bf16_kernel<<<(D * D) / 1024, 256, 0, stream>>>(Wv, Wv_b, D * D);
    cvt_f32_to_bf16_kernel<<<(D * D) / 1024, 256, 0, stream>>>(Wo, Wo_b, D * D);

    const float invSqrtD = 0.03125f;  // 1/sqrt(1024)
    auto gemmBlocks = [](int M, int N) { return ((M / 64) * (N / 64)) / 8; };

    // ---- 2. projections ----
    // Q = (target @ Wq^T + bq) * invSqrtD  -> bf16 [MA, D]
    gemm_bf16_kernel<1><<<gemmBlocks(MA, D), 256, 0, stream>>>(
        tgt_b, D, Wq_b, D, Qb, D, bq, invSqrtD, MA, D, D, T, 0LL);
    // K = source @ Wk^T + bk -> bf16 [MA, D]
    gemm_bf16_kernel<1><<<gemmBlocks(MA, D), 256, 0, stream>>>(
        src_b, D, Wk_b, D, Kb, D, bk, 1.0f, MA, D, D, T, 0LL);
    // V = source @ Wv^T + bv, stored transposed per batch: Vt[b][e][s]
    gemm_bf16_kernel<2><<<gemmBlocks(MA, D), 256, 0, stream>>>(
        src_b, D, Wv_b, D, Vt, T, bv, 1.0f, MA, D, D, T, (long long)D * T);

    // ---- 3. per-batch attention (S buffer stays L2-resident: 16MB << 192MB) ----
    for (int b = 0; b < B; ++b) {
        const bf16_t* Qbb = Qb + (size_t)b * T * D;
        const bf16_t* Kbb = Kb + (size_t)b * T * D;
        const bf16_t* Vtb = Vt + (size_t)b * D * T;
        bf16_t*       Obb = Ob + (size_t)b * T * D;

        // S = Q @ K^T   (scale already folded into Q)  f32 [T, T]
        gemm_bf16_kernel<0><<<gemmBlocks(T, T), 256, 0, stream>>>(
            Qbb, D, Kbb, D, Sb, T, nullptr, 1.0f, T, T, D, T, 0LL);

        // P = softmax(S) -> bf16 [T, T]
        softmax_bf16_kernel<<<T, 256, 0, stream>>>(Sb, Pb, T);

        // O = P @ V  (Bcol[n=e][k=s] = Vt[e][s]) -> bf16 [T, D]
        gemm_bf16_kernel<1><<<gemmBlocks(T, D), 256, 0, stream>>>(
            Pb, T, Vtb, T, Obb, D, nullptr, 1.0f, T, D, T, T, 0LL);
    }

    // ---- 4. out = O @ Wo^T + bo -> f32 [MA, D] ----
    gemm_bf16_kernel<0><<<gemmBlocks(MA, D), 256, 0, stream>>>(
        Ob, D, Wo_b, D, out, D, bo, 1.0f, MA, D, D, T, 0LL);
}
